// TextEncoder_14869176779197
// MI455X (gfx1250) — compile-verified, matrix operand order checked by reading, not run on hardware
//
#include <hip/hip_runtime.h>
#include <cstdint>
#include <cstddef>

// ---------------- problem constants ----------------
#define BB   512
#define LL   77
#define DD   512
#define HH   8
#define HDD  64
#define NLAYER 12
#define MM   (BB * LL)        // 39424 rows (multiple of 64)
#define P_   1
#define C_   10
#define PC_  11
#define MW_  0.1f
#define NEG_ (-1e30f)

#ifndef USE_TDM
#define USE_TDM 1
#endif

typedef __bf16 bf16;
typedef bf16  v8bf  __attribute__((ext_vector_type(8)));
typedef bf16  v16bf __attribute__((ext_vector_type(16)));
typedef float v8f   __attribute__((ext_vector_type(8)));
typedef unsigned int u32x4 __attribute__((ext_vector_type(4)));
typedef int   i32x4 __attribute__((ext_vector_type(4)));
typedef int   i32x8 __attribute__((ext_vector_type(8)));

// ---------------- device helpers ----------------
__device__ __forceinline__ v8f wmma_bf16(v16bf a, v16bf b, v8f c) {
  return __builtin_amdgcn_wmma_f32_16x16x32_bf16(false, a, false, b, (short)0, c,
                                                 false, false);
}

// ISA 16-bit A/B fragment: lane holds k = {kb..kb+7, kb+16..kb+23}, kb=(lane>=16)?8:0
__device__ __forceinline__ v16bf frag_from_row(const bf16* rowPtr, int kb) {
  v8bf lo = *(const v8bf*)(rowPtr + kb);
  v8bf hi = *(const v8bf*)(rowPtr + kb + 16);
  return __builtin_shufflevector(lo, hi, 0, 1, 2, 3, 4, 5, 6, 7,
                                 8, 9, 10, 11, 12, 13, 14, 15);
}

#if USE_TDM
// Pack a 2-D Tensor-DMA descriptor (CDNA5 ISA) and issue a
// TENSOR_LOAD_TO_LDS. Tile = tile_rows x tile_k bf16 elements, row-major,
// row stride = stride_elems. TENSORcnt is per-wave; wave 0 drives the DMA.
__device__ __forceinline__ void tdm_load_tile(unsigned lds_addr, const void* gptr,
                                              unsigned tile_k, unsigned tile_rows,
                                              unsigned tensor_k, unsigned tensor_rows,
                                              unsigned stride_elems) {
  unsigned long long ga = (unsigned long long)gptr;
  u32x4 g0 = {
      1u,                                         // count=1, user-mode load
      lds_addr,                                   // LDS byte address
      (unsigned)(ga & 0xFFFFFFFFu),               // global addr [31:0]
      (unsigned)((ga >> 32) & 0x01FFFFFFu) | (2u << 30) // addr[56:32] | type=2
  };
  i32x8 g1;
  g1[0] = (1 << 16);                                            // data_size=1 (2B)
  g1[1] = (int)((tensor_k & 0xFFFFu) << 16);                    // tensor_dim0 lo16
  g1[2] = (int)(((tensor_k >> 16) & 0xFFFFu) |
                ((tensor_rows & 0xFFFFu) << 16));               // dim0 hi | dim1 lo
  g1[3] = (int)(((tensor_rows >> 16) & 0xFFFFu) |
                ((tile_k & 0xFFFFu) << 16));                    // dim1 hi | tile_dim0
  g1[4] = (int)(tile_rows & 0xFFFFu);                           // tile_dim1, tile_dim2=0
  g1[5] = (int)stride_elems;                                    // dim0_stride lo32
  g1[6] = 0;                                                    // dim0_stride hi16
  g1[7] = 0;
  i32x4 z4 = {0, 0, 0, 0};
  i32x8 z8 = {0, 0, 0, 0, 0, 0, 0, 0};
  __builtin_amdgcn_tensor_load_to_lds(g0, g1, z4, z4, z8, 0);
}
#endif

// additive attention mask, matching build_text_mask()
__device__ __forceinline__ float mask_val(int i, int j, int len) {
  float m = (j > i) ? NEG_ : 0.0f;
  bool ctx_row = (i >= P_) && (i < PC_);
  if (ctx_row && (j < len)) m = 0.0f;
  if ((i >= PC_) && (i < len) && (j >= P_) && (j < PC_)) m = NEG_;
  if (ctx_row && (j >= P_) && (j < PC_)) m = NEG_;
  if (j >= len) m = NEG_;
  bool eot = (i == len - 1);
  if (eot && (j < PC_)) m = 0.0f;
  if (eot && (j >= PC_) && (j < len)) m = MW_;
  return m;
}

// ---------------- elementwise kernels ----------------
__global__ void f32_to_bf16_kernel(const float* __restrict__ in,
                                   bf16* __restrict__ out, size_t n) {
  size_t i = (size_t)blockIdx.x * blockDim.x + threadIdx.x;
  if (i < n) out[i] = (bf16)in[i];
}

// text_projection (D,PROJ) -> bf16 (PROJ,D) so GEMM sees (N,K)
__global__ void transpose_cvt_kernel(const float* __restrict__ in,
                                     bf16* __restrict__ out, int rows, int cols) {
  int i = blockIdx.x * 256 + threadIdx.x;
  if (i < rows * cols) {
    int r = i / cols, c = i % cols;
    out[(size_t)c * rows + r] = (bf16)in[i];
  }
}

__global__ void embed_kernel(const float* __restrict__ prompts,
                             const float* __restrict__ pos,
                             float* __restrict__ x, size_t n) {
  size_t i = (size_t)blockIdx.x * blockDim.x + threadIdx.x;
  if (i < n) x[i] = prompts[i] + pos[i % (size_t)(LL * DD)];
}

// ---------------- layernorm: f32 row -> bf16 row ----------------
__global__ void ln_rows_kernel(const float* __restrict__ x,
                               const float* __restrict__ w,
                               const float* __restrict__ b,
                               bf16* __restrict__ out) {
  int row = blockIdx.x;
  const float* xr = x + (size_t)row * DD;
  int t = threadIdx.x, lane = t & 31, wid = t >> 5;
  float v0 = xr[t], v1 = xr[t + 256];
  __shared__ float red[8];
  __shared__ float mu_s, rs_s;
  float s = v0 + v1;
  for (int o = 16; o; o >>= 1) s += __shfl_xor(s, o, 32);
  if (lane == 0) red[wid] = s;
  __syncthreads();
  if (t == 0) {
    float tot = 0.f;
    for (int i = 0; i < 8; ++i) tot += red[i];
    mu_s = tot * (1.0f / DD);
  }
  __syncthreads();
  float mu = mu_s;
  float d0 = v0 - mu, d1 = v1 - mu;
  float q = d0 * d0 + d1 * d1;
  for (int o = 16; o; o >>= 1) q += __shfl_xor(q, o, 32);
  if (lane == 0) red[wid] = q;
  __syncthreads();
  if (t == 0) {
    float tot = 0.f;
    for (int i = 0; i < 8; ++i) tot += red[i];
    rs_s = rsqrtf(tot * (1.0f / DD) + 1e-5f);
  }
  __syncthreads();
  float rs = rs_s;
  out[(size_t)row * DD + t]       = (bf16)(d0 * rs * w[t] + b[t]);
  out[(size_t)row * DD + t + 256] = (bf16)(d1 * rs * w[t + 256] + b[t + 256]);
}

// final LN + EOT gather -> bf16 (B, D)
__global__ void lnf_gather_kernel(const float* __restrict__ x,
                                  const int* __restrict__ tokens,
                                  const float* __restrict__ w,
                                  const float* __restrict__ b,
                                  bf16* __restrict__ g) {
  int bi = blockIdx.x;
  __shared__ int eot_s;
  int t = threadIdx.x, lane = t & 31, wid = t >> 5;
  if (t == 0) {
    int best = tokens[bi * LL], idx = 0;
    for (int i = 1; i < LL; ++i) {
      int v = tokens[bi * LL + i];
      if (v > best) { best = v; idx = i; }
    }
    eot_s = idx;
  }
  __syncthreads();
  const float* xr = x + ((size_t)bi * LL + eot_s) * DD;
  float v0 = xr[t], v1 = xr[t + 256];
  __shared__ float red[8];
  __shared__ float mu_s, rs_s;
  float s = v0 + v1;
  for (int o = 16; o; o >>= 1) s += __shfl_xor(s, o, 32);
  if (lane == 0) red[wid] = s;
  __syncthreads();
  if (t == 0) {
    float tot = 0.f;
    for (int i = 0; i < 8; ++i) tot += red[i];
    mu_s = tot * (1.0f / DD);
  }
  __syncthreads();
  float mu = mu_s;
  float d0 = v0 - mu, d1 = v1 - mu;
  float q = d0 * d0 + d1 * d1;
  for (int o = 16; o; o >>= 1) q += __shfl_xor(q, o, 32);
  if (lane == 0) red[wid] = q;
  __syncthreads();
  if (t == 0) {
    float tot = 0.f;
    for (int i = 0; i < 8; ++i) tot += red[i];
    rs_s = rsqrtf(tot * (1.0f / DD) + 1e-5f);
  }
  __syncthreads();
  float rs = rs_s;
  g[(size_t)bi * DD + t]       = (bf16)(d0 * rs * w[t] + b[t]);
  g[(size_t)bi * DD + t + 256] = (bf16)(d1 * rs * w[t + 256] + b[t + 256]);
}

// ---------------- WMMA GEMM: C[m,n] = sum_k A[m,k]*W[n,k] (+epilogue) ----------
// Block tile 64(M) x 128(N), 8 waves (4x2), each wave 16x64 = 4 accumulators.
// K staged 64-wide, TDM double-buffered: stage s+1 DMA overlaps stage s math.
// MODE 0: +bias -> f32 out     MODE 1: +bias +residual -> f32 out
// MODE 2: +bias, GELU -> bf16  MODE 3: +bias -> bf16 out
template <int MODE>
__global__ __launch_bounds__(256) void gemm_wmma(
    const bf16* __restrict__ A, const bf16* __restrict__ W,
    const float* __restrict__ bias, const float* __restrict__ res,
    float* __restrict__ outF, bf16* __restrict__ outB, int M, int N, int K) {
  __shared__ bf16 sA[2 * 64 * 64];    // 16 KB, double buffered
  __shared__ bf16 sW[2 * 128 * 64];   // 32 KB, double buffered
  int tid = threadIdx.x, lane = tid & 31, wave = tid >> 5;
  int wm = wave >> 1, wn = wave & 1;  // 4 x 2 waves
  int blockM = blockIdx.y * 64, blockN = blockIdx.x * 128;
  int kb = (lane >> 4) * 8;
  v8f acc[4];
#pragma unroll
  for (int cc = 0; cc < 4; ++cc) acc[cc] = (v8f){0, 0, 0, 0, 0, 0, 0, 0};

  const int nStages = K >> 6;         // 64-wide K stages

#if USE_TDM
  if (tid < 32) {                     // prologue: stage 0 into buffer 0
    tdm_load_tile((unsigned)(size_t)(sA), A + (size_t)blockM * K,
                  64u, 64u, (unsigned)K, (unsigned)M, (unsigned)K);
    tdm_load_tile((unsigned)(size_t)(sW), W + (size_t)blockN * K,
                  64u, 128u, (unsigned)K, (unsigned)N, (unsigned)K);
  }
#endif

  for (int s = 0; s < nStages; ++s) {
    int cur = s & 1;
#if USE_TDM
    if (tid < 32) __builtin_amdgcn_s_wait_tensorcnt(0);  // buf[cur] landed
    __syncthreads();                                     // publish to all waves
    if ((tid < 32) && (s + 1 < nStages)) {               // prefetch stage s+1
      int k1 = (s + 1) << 6;
      tdm_load_tile((unsigned)(size_t)(sA + (1 - cur) * 64 * 64),
                    A + (size_t)blockM * K + k1,
                    64u, 64u, (unsigned)K, (unsigned)M, (unsigned)K);
      tdm_load_tile((unsigned)(size_t)(sW + (1 - cur) * 128 * 64),
                    W + (size_t)blockN * K + k1,
                    64u, 128u, (unsigned)K, (unsigned)N, (unsigned)K);
    }
#else
    __syncthreads();
    {
      int k0 = s << 6;
      // A: 64x64 = 512 v8 chunks; W: 128x64 = 1024 v8 chunks
#pragma unroll
      for (int it = 0; it < 2; ++it) {
        int c = tid + it * 256, r = c >> 3, col = (c & 7) * 8;
        *(v8bf*)(sA + cur * 64 * 64 + r * 64 + col) =
            *(const v8bf*)(A + (size_t)(blockM + r) * K + k0 + col);
      }
#pragma unroll
      for (int it = 0; it < 4; ++it) {
        int c = tid + it * 256, r = c >> 3, col = (c & 7) * 8;
        *(v8bf*)(sW + cur * 128 * 64 + r * 64 + col) =
            *(const v8bf*)(W + (size_t)(blockN + r) * K + k0 + col);
      }
    }
    __syncthreads();
#endif
    const bf16* arow = sA + cur * 64 * 64 + (wm * 16 + (lane & 15)) * 64;
    const bf16* brow = sW + cur * 128 * 64 + (wn * 64 + (lane & 15)) * 64;
#pragma unroll
    for (int kc = 0; kc < 64; kc += 32) {
      v16bf af = frag_from_row(arow, kc + kb);
#pragma unroll
      for (int cc = 0; cc < 4; ++cc) {
        v16bf bf = frag_from_row(brow + cc * 16 * 64, kc + kb);
        acc[cc] = wmma_bf16(af, bf, acc[cc]);
      }
    }
#if USE_TDM
    __syncthreads();   // all waves done with buf[cur] before it is re-filled
#endif
  }

  int mBase = blockM + wm * 16 + (lane >> 4) * 8;
#pragma unroll
  for (int cc = 0; cc < 4; ++cc) {
    int n = blockN + wn * 64 + cc * 16 + (lane & 15);
    float bv = bias ? bias[n] : 0.0f;
#pragma unroll
    for (int r = 0; r < 8; ++r) {
      int m = mBase + r;
      float v = acc[cc][r] + bv;
      if (MODE == 1) v += res[(size_t)m * N + n];
      if (MODE == 2) v = v / (1.0f + __expf(-1.702f * v));
      if (MODE <= 1) outF[(size_t)m * N + n] = v;
      else           outB[(size_t)m * N + n] = (bf16)v;
    }
  }
}

// ---------------- fused masked attention, one block per (b,h) --------------
// qkv: bf16 (M, 1536) [q|k|v]; o: bf16 (M, 512)
__global__ __launch_bounds__(256) void attention_kernel(
    const bf16* __restrict__ qkv, const int* __restrict__ tokens,
    bf16* __restrict__ o) {
  extern __shared__ char smem[];
  bf16*  sK = (bf16*)smem;                 // 96 x 64
  bf16*  sV = sK + 96 * 64;                // 96 x 64
  float* sS = (float*)(sV + 96 * 64);      // 80 x 96 (f32 scores)
  bf16*  sP = (bf16*)(sS + 80 * 96);       // 80 x 96 (bf16 probs)
  __shared__ int lenS;

  int bh = blockIdx.x;
  int b = bh / HH, h = bh % HH;
  int tid = threadIdx.x, lane = tid & 31, wave = tid >> 5;
  int kb = (lane >> 4) * 8;

  if (tid == 0) {
    int best = tokens[b * LL], idx = 0;
    for (int i = 1; i < LL; ++i) {
      int v = tokens[b * LL + i];
      if (v > best) { best = v; idx = i; }
    }
    lenS = idx + 1;
  }

  // stage K and V (zero-padded to 96 rows)
  for (int c = tid; c < 96 * 64 / 8; c += 256) {
    int j = c >> 3, d8 = (c & 7) * 8;
    v8bf kk, vv;
    if (j < LL) {
      const bf16* base = qkv + ((size_t)b * LL + j) * 1536 + (size_t)h * 64 + d8;
      kk = *(const v8bf*)(base + 512);
      vv = *(const v8bf*)(base + 1024);
    } else {
#pragma unroll
      for (int e = 0; e < 8; ++e) { kk[e] = (bf16)0.0f; vv[e] = (bf16)0.0f; }
    }
    *(v8bf*)(sK + j * 64 + d8) = kk;
    *(v8bf*)(sV + j * 64 + d8) = vv;
  }
  __syncthreads();
  int len = lenS;

  // S = Q K^T * scale + mask   (5 i-tiles x 6 j-tiles, K=64 -> 2 WMMAs)
  for (int t = wave; t < 30; t += 8) {
    int ti = t / 6, tj = t % 6;
    int gi = ti * 16 + (lane & 15);
    int qi = gi < LL ? gi : LL - 1;
    const bf16* qrow = qkv + ((size_t)b * LL + qi) * 1536 + (size_t)h * 64;
    const bf16* krow = sK + (tj * 16 + (lane & 15)) * 64;
    v8f c = {0, 0, 0, 0, 0, 0, 0, 0};
    c = wmma_bf16(frag_from_row(qrow, kb),      frag_from_row(krow, kb),      c);
    c = wmma_bf16(frag_from_row(qrow, 32 + kb), frag_from_row(krow, 32 + kb), c);
    int jj = tj * 16 + (lane & 15);
    int mrow = ti * 16 + (lane >> 4) * 8;
#pragma unroll
    for (int r = 0; r < 8; ++r)
      sS[(mrow + r) * 96 + jj] = c[r] * 0.125f + mask_val(mrow + r, jj, len);
  }
  __syncthreads();

  // row softmax (wave32 shuffle reduction), write bf16 probabilities
  for (int r = wave; r < 80; r += 8) {
    float v0 = sS[r * 96 + lane];
    float v1 = sS[r * 96 + lane + 32];
    float v2 = sS[r * 96 + lane + 64];
    float mx = fmaxf(v0, fmaxf(v1, v2));
    for (int od = 16; od; od >>= 1) mx = fmaxf(mx, __shfl_xor(mx, od, 32));
    float e0 = __expf(v0 - mx), e1 = __expf(v1 - mx), e2 = __expf(v2 - mx);
    float sm = e0 + e1 + e2;
    for (int od = 16; od; od >>= 1) sm += __shfl_xor(sm, od, 32);
    float inv = 1.0f / sm;
    sP[r * 96 + lane]      = (bf16)(e0 * inv);
    sP[r * 96 + lane + 32] = (bf16)(e1 * inv);
    sP[r * 96 + lane + 64] = (bf16)(e2 * inv);
  }
  __syncthreads();

  // O = P V  (5 i-tiles x 4 d-tiles, K padded to 96 -> 3 WMMAs)
  for (int t = wave; t < 20; t += 8) {
    int ti = t / 4, td = t % 4;
    const bf16* prow = sP + (ti * 16 + (lane & 15)) * 96;
    int n = td * 16 + (lane & 15);
    v8f c = {0, 0, 0, 0, 0, 0, 0, 0};
#pragma unroll
    for (int kc = 0; kc < 96; kc += 32) {
      v16bf a = frag_from_row(prow, kc + kb);
      v16bf bb;
#pragma unroll
      for (int e = 0; e < 8; ++e) {
        bb[e]     = sV[(kc + kb + e) * 64 + n];
        bb[e + 8] = sV[(kc + kb + 16 + e) * 64 + n];
      }
      c = wmma_bf16(a, bb, c);
    }
    int mrow = ti * 16 + (lane >> 4) * 8;
#pragma unroll
    for (int r = 0; r < 8; ++r) {
      int gi = mrow + r;
      if (gi < LL)
        o[((size_t)b * LL + gi) * DD + (size_t)h * 64 + n] = (bf16)c[r];
    }
  }
}

// ---------------- host orchestration ----------------
static inline void launch_gemm(int mode, const bf16* A, const bf16* W,
                               const float* bias, const float* res, float* of,
                               bf16* ob, int M, int N, int K, hipStream_t s) {
  dim3 g(N / 128, M / 64), blk(256);
  switch (mode) {
    case 0: gemm_wmma<0><<<g, blk, 0, s>>>(A, W, bias, res, of, ob, M, N, K); break;
    case 1: gemm_wmma<1><<<g, blk, 0, s>>>(A, W, bias, res, of, ob, M, N, K); break;
    case 2: gemm_wmma<2><<<g, blk, 0, s>>>(A, W, bias, res, of, ob, M, N, K); break;
    default: gemm_wmma<3><<<g, blk, 0, s>>>(A, W, bias, res, of, ob, M, N, K); break;
  }
}

extern "C" void kernel_launch(void* const* d_in, const int* in_sizes, int n_in,
                              void* d_out, int out_size, void* d_ws, size_t ws_size,
                              hipStream_t stream) {
  (void)in_sizes; (void)n_in; (void)out_size; (void)ws_size;
  const float* prompts = (const float*)d_in[0];
  const float* pos_emb = (const float*)d_in[1];
  const int*   tokens  = (const int*)d_in[2];
  const float* ln1_w = (const float*)d_in[3];
  const float* ln1_b = (const float*)d_in[4];
  const float* inw   = (const float*)d_in[5];
  const float* inb   = (const float*)d_in[6];
  const float* ow    = (const float*)d_in[7];
  const float* ob    = (const float*)d_in[8];
  const float* ln2_w = (const float*)d_in[9];
  const float* ln2_b = (const float*)d_in[10];
  const float* fw    = (const float*)d_in[11];
  const float* fb    = (const float*)d_in[12];
  const float* cw    = (const float*)d_in[13];
  const float* cb    = (const float*)d_in[14];
  const float* lnf_w = (const float*)d_in[15];
  const float* lnf_b = (const float*)d_in[16];
  const float* tproj = (const float*)d_in[17];

  // ---- workspace carve-out ----
  char* ws = (char*)d_ws;
  size_t off = 0;
  auto carve = [&](size_t bytes) {
    void* p = ws + off;
    off = (off + bytes + 255) & ~(size_t)255;
    return p;
  };
  const size_t n_inw = (size_t)NLAYER * 3 * DD * DD;
  const size_t n_ow  = (size_t)NLAYER * DD * DD;
  const size_t n_fw  = (size_t)NLAYER * 4 * DD * DD;
  const size_t n_cw  = (size_t)NLAYER * DD * 4 * DD;
  const size_t n_tp  = (size_t)DD * DD;
  bf16* inw_b = (bf16*)carve(n_inw * 2);
  bf16* ow_b  = (bf16*)carve(n_ow * 2);
  bf16* fw_b  = (bf16*)carve(n_fw * 2);
  bf16* cw_b  = (bf16*)carve(n_cw * 2);
  bf16* tp_b  = (bf16*)carve(n_tp * 2);
  float* x    = (float*)carve((size_t)MM * DD * 4);
  bf16* hbuf  = (bf16*)carve((size_t)MM * DD * 2);
  bf16* qkvb  = (bf16*)carve((size_t)MM * 3 * DD * 2);
  bf16* obuf  = (bf16*)carve((size_t)MM * DD * 2);
  bf16* a4    = (bf16*)carve((size_t)MM * 4 * DD * 2);
  bf16* gbuf  = (bf16*)carve((size_t)BB * DD * 2);

  // ---- one-time weight conversion (idempotent) ----
  auto cvt = [&](const float* src, bf16* dst, size_t n) {
    f32_to_bf16_kernel<<<(unsigned)((n + 255) / 256), 256, 0, stream>>>(src, dst, n);
  };
  cvt(inw, inw_b, n_inw);
  cvt(ow, ow_b, n_ow);
  cvt(fw, fw_b, n_fw);
  cvt(cw, cw_b, n_cw);
  transpose_cvt_kernel<<<(n_tp + 255) / 256, 256, 0, stream>>>(tproj, tp_b, DD, DD);

  // ---- embed ----
  {
    size_t n = (size_t)MM * DD;
    embed_kernel<<<(unsigned)((n + 255) / 256), 256, 0, stream>>>(prompts, pos_emb, x, n);
  }

  const size_t attn_smem = (size_t)96 * 64 * 2 * 2 + (size_t)80 * 96 * 4 +
                           (size_t)80 * 96 * 2;   // 70656 B

  // ---- transformer layers ----
  for (int l = 0; l < NLAYER; ++l) {
    const bf16* linw = inw_b + (size_t)l * 3 * DD * DD;
    const bf16* low  = ow_b  + (size_t)l * DD * DD;
    const bf16* lfw  = fw_b  + (size_t)l * 4 * DD * DD;
    const bf16* lcw  = cw_b  + (size_t)l * DD * 4 * DD;

    ln_rows_kernel<<<MM, 256, 0, stream>>>(x, ln1_w + l * DD, ln1_b + l * DD, hbuf);
    launch_gemm(3, hbuf, linw, inb + (size_t)l * 3 * DD, nullptr, nullptr, qkvb,
                MM, 3 * DD, DD, stream);
    attention_kernel<<<BB * HH, 256, attn_smem, stream>>>(qkvb, tokens, obuf);
    launch_gemm(1, obuf, low, ob + (size_t)l * DD, x, x, nullptr,
                MM, DD, DD, stream);
    ln_rows_kernel<<<MM, 256, 0, stream>>>(x, ln2_w + l * DD, ln2_b + l * DD, hbuf);
    launch_gemm(2, hbuf, lfw, fb + (size_t)l * 4 * DD, nullptr, nullptr, a4,
                MM, 4 * DD, DD, stream);
    launch_gemm(1, a4, lcw, cb + (size_t)l * DD, x, x, nullptr,
                MM, DD, 4 * DD, stream);
  }

  // ---- final LN + EOT gather + projection ----
  lnf_gather_kernel<<<BB, 256, 0, stream>>>(x, tokens, lnf_w, lnf_b, gbuf);
  launch_gemm(0, gbuf, tp_b, nullptr, nullptr, (float*)d_out, nullptr,
              BB, DD, DD, stream);
}
